// MultiheadAttention_41601053229405
// MI455X (gfx1250) — matrix operand order repaired and checked by reading, not compiled, not run on hardware
//
#include <hip/hip_runtime.h>

// ---------------------------------------------------------------------------
// MI455X / gfx1250 multihead attention, bf16 WMMA + big-LDS fusion +
// async global->LDS staging (ASYNCcnt path) for pure-copy tiles.
// ---------------------------------------------------------------------------

typedef unsigned short u16;
typedef unsigned int   u32;
typedef __attribute__((ext_vector_type(16))) __bf16 v16bf;
typedef __attribute__((ext_vector_type(8)))  float  v8f;
typedef __attribute__((ext_vector_type(4)))  u32    u32x4;
typedef __attribute__((ext_vector_type(2)))  u32    u32x2;

constexpr int Bc = 4, NTc = 1024, NSc = 1024, Dc = 1024, Hc = 16, HDc = 64;

// Attention kernel LDS plan (dynamic; ~282 KB of the WGP's 320 KB).
constexpr int TM   = 32;    // query rows per workgroup
constexpr int QST  = 72;    // qtile halfword stride  (144B rows, 16B aligned, bank-spread)
constexpr int KST  = 72;    // k-head halfword stride
constexpr int SST  = 1028;  // scores f32 stride (+4 kills the 8-row bank collision)
constexpr int VST  = 40;    // v-tile halfword stride (80B rows)
constexpr int WST  = 1032;  // weights-tile halfword stride (2064B rows)
constexpr size_t OFF_SC = (size_t)TM * QST * 2;            // after qtile
constexpr size_t OFF_VT = OFF_SC + (size_t)TM * SST * 4;   // after scores
constexpr size_t OFF_KT = OFF_VT + (size_t)HDc * VST * 2;  // after vtile
constexpr size_t ATTN_SMEM = OFF_KT + (size_t)NSc * KST * 2;  // = 288768 B

__device__ __forceinline__ u16 f2bf(float f) {        // f32 -> bf16 RNE
  u32 u = __float_as_uint(f);
  u += 0x7FFFu + ((u >> 16) & 1u);
  return (u16)(u >> 16);
}
__device__ __forceinline__ float bf2f(u16 v) { return __uint_as_float(((u32)v) << 16); }

// Generic pointers into __shared__ carry the LDS byte address in their low
// 32 bits.
__device__ __forceinline__ u32 lds_addr(const void* p) { return (u32)(size_t)p; }

// Async global->LDS copies (VGLOBAL "GV" mode, tracked with ASYNCcnt).
__device__ __forceinline__ void async_copy_b128(u32 lds_byte, const void* g) {
  asm volatile("global_load_async_to_lds_b128 %0, %1, off"
               :: "v"(lds_byte), "v"(g) : "memory");
}
__device__ __forceinline__ void async_copy_b64(u32 lds_byte, const void* g) {
  asm volatile("global_load_async_to_lds_b64 %0, %1, off"
               :: "v"(lds_byte), "v"(g) : "memory");
}
__device__ __forceinline__ void wait_async0() {
  asm volatile("s_wait_asynccnt 0" ::: "memory");
}

union Frag { v16bf v; u32x4 q[2]; };

// A fragment (16x32 bf16, M=lane&15): elements j -> K = j + 8*(half + (j>=8)).
// Two 16B LDS loads at halfword offsets k0+8*half and k0+16+8*half.
__device__ __forceinline__ v16bf ld_frag_a(const u16* tile, int row, int stride, int k0) {
  int lane = threadIdx.x & 31, r = lane & 15, hh = lane >> 4;
  const u16* p = tile + (size_t)(row + r) * stride + k0 + 8 * hh;
  Frag f;
  f.q[0] = *(const u32x4*)p;
  f.q[1] = *(const u32x4*)(p + 16);
  return f.v;
}
// B fragment (32x16 bf16, N=lane&15): elements j -> K = j + 8*(half + (j>=8)).
// Tile stored transposed as [n][k] so each lane does two 16B loads, as for A.
__device__ __forceinline__ v16bf ld_frag_b(const u16* tile, int col, int stride, int k0) {
  int lane = threadIdx.x & 31, r = lane & 15, hh = lane >> 4;
  const u16* p = tile + (size_t)(col + r) * stride + k0 + 8 * hh;
  Frag f;
  f.q[0] = *(const u32x4*)p;
  f.q[1] = *(const u32x4*)(p + 16);
  return f.v;
}
__device__ __forceinline__ v8f wmma_bf16(v16bf a, v16bf b, v8f c) {
  return __builtin_amdgcn_wmma_f32_16x16x32_bf16(false, a, false, b, (short)0, c, false, false);
}

// ---------------------------------------------------------------------------
// GEMM: Out[M,N] = A[M,K] @ W[N,K]^T + bias[N]   (torch f.linear)
// Block tile 128x64, BK=32 (one WMMA K-step). 8 waves = 4(M) x 2(N),
// each wave owns a 32x32 tile (2x2 WMMA accumulators).
// A_BF16 path stages its tile with async global->LDS copies.
// ---------------------------------------------------------------------------
template <bool A_BF16, bool OUT_BF16>
__global__ __launch_bounds__(256) void gemm_xwT(const void* __restrict__ Ain,
                                                const float* __restrict__ W,
                                                const float* __restrict__ bias,
                                                void* __restrict__ Out,
                                                int M, int N, int K) {
  constexpr int BM = 128, BN = 64, BK = 32, LST = BK + 8;  // 80B rows
  __shared__ u16 As[BM * LST];  // 10 KB
  __shared__ u16 Bs[BN * LST];  // 5 KB

  const int tid = threadIdx.x;
  const int m0 = blockIdx.y * BM, n0 = blockIdx.x * BN;
  const int wave = tid >> 5, lane = tid & 31;
  const int wm = wave >> 1, wn = wave & 1;
  const int r = lane & 15, hh = lane >> 4;

  const float* Af = (const float*)Ain;
  const u16*   Ab = (const u16*)Ain;

  v8f acc[4] = {};

  for (int k0 = 0; k0 < K; k0 += BK) {
    if (k0) __syncthreads();  // prior-iteration frag reads must finish before restage
    // ---- stage A tile (128x32) ----
#pragma unroll
    for (int i = 0; i < 4; i++) {
      int idx = tid + 256 * i, row = idx >> 3, c4 = (idx & 7) << 2;
      if (A_BF16) {
        // bf16 -> bf16: pure copy, use the async DMA path (no VGPR round-trip)
        async_copy_b64(lds_addr(As + row * LST + c4),
                       Ab + (size_t)(m0 + row) * K + k0 + c4);
      } else {
        const float* s = Af + (size_t)(m0 + row) * K + k0 + c4;
        __builtin_prefetch(s + BK, 0, 1);  // global_prefetch for next K-slab
        u32x2 pk;
        pk.x = (u32)f2bf(s[0]) | ((u32)f2bf(s[1]) << 16);
        pk.y = (u32)f2bf(s[2]) | ((u32)f2bf(s[3]) << 16);
        *(u32x2*)(As + row * LST + c4) = pk;
      }
    }
    // ---- stage B tile: rows of W (W[n][k], k contiguous) -> Bs[n][k] ----
#pragma unroll
    for (int i = 0; i < 2; i++) {
      int idx = tid + 256 * i, row = idx >> 3, c4 = (idx & 7) << 2;
      const float* s = W + (size_t)(n0 + row) * K + k0 + c4;
      u32x2 pk;
      pk.x = (u32)f2bf(s[0]) | ((u32)f2bf(s[1]) << 16);
      pk.y = (u32)f2bf(s[2]) | ((u32)f2bf(s[3]) << 16);
      *(u32x2*)(Bs + row * LST + c4) = pk;
    }
    if (A_BF16) wait_async0();  // async LDS writes visible before barrier
    __syncthreads();

    v16bf a0 = ld_frag_a(As, wm * 32,      LST, 0);
    v16bf a1 = ld_frag_a(As, wm * 32 + 16, LST, 0);
    v16bf b0 = ld_frag_b(Bs, wn * 32,      LST, 0);
    v16bf b1 = ld_frag_b(Bs, wn * 32 + 16, LST, 0);
    acc[0] = wmma_bf16(a0, b0, acc[0]);
    acc[1] = wmma_bf16(a0, b1, acc[1]);
    acc[2] = wmma_bf16(a1, b0, acc[2]);
    acc[3] = wmma_bf16(a1, b1, acc[3]);
  }

  // ---- epilogue: + bias, write f32 or bf16 ----
#pragma unroll
  for (int t = 0; t < 4; t++) {
    int rt = t >> 1, ct = t & 1;
    int col = n0 + wn * 32 + ct * 16 + r;
    float bv = bias[col];
#pragma unroll
    for (int i = 0; i < 8; i++) {
      int row = m0 + wm * 32 + rt * 16 + i + 8 * hh;
      float o = acc[t][i] + bv;
      if (OUT_BF16)
        ((u16*)Out)[(size_t)row * N + col] = f2bf(o);
      else
        ((float*)Out)[(size_t)row * N + col] = o;
    }
  }
}

// ---------------------------------------------------------------------------
// Fused attention per (b, h, 32-row q tile):
//   scores = (q*scale) @ k^T  (bf16 WMMA, f32 scores in LDS)
//   + mask, softmax over full NS row, write f32 weights to d_out,
//   keep bf16 weights in LDS, then ctx = P @ V (bf16 WMMA) -> bf16 ctx.
// Whole K-head (1024x64 bf16 = 144 KB, staged via async DMA) + 32x1024 f32
// scores live in LDS.
// ---------------------------------------------------------------------------
__global__ __launch_bounds__(256) void attn_fused(const u16* __restrict__ qp,
                                                  const u16* __restrict__ kp,
                                                  const u16* __restrict__ vp,
                                                  const float* __restrict__ mask,
                                                  float* __restrict__ wout,
                                                  u16* __restrict__ ctx) {
  extern __shared__ __align__(16) char smem[];
  u16*   qt = (u16*)smem;                 // [TM][QST]
  float* sc = (float*)(smem + OFF_SC);    // [TM][SST]
  u16*   vt = (u16*)(smem + OFF_VT);      // [HD][VST]  (transposed v chunk)
  u16*   kt = (u16*)(smem + OFF_KT);      // [NS][KST]
  u16*   wt = kt;                         // [TM][WST]  bf16 softmax weights (aliases kt)

  const int blk = blockIdx.x;
  const int ttile = blk & 31, h = (blk >> 5) & 15, b = blk >> 9;
  const int t0 = ttile * TM;
  const int tid = threadIdx.x, wave = tid >> 5, lane = tid & 31;
  const int r = lane & 15, hh = lane >> 4;

  // ---- stage entire K head via async DMA: kt[s][d] (pure bf16 copy) ----
#pragma unroll 4
  for (int i = 0; i < 32; i++) {
    int idx = tid + 256 * i, s = idx >> 3, c8 = (idx & 7) << 3;
    async_copy_b128(lds_addr(kt + (size_t)s * KST + c8),
                    kp + (size_t)(b * NSc + s) * Dc + h * HDc + c8);
  }
  // ---- stage q tile (pre-scaled by 1/sqrt(HD) = 0.125) while DMA runs ----
#pragma unroll
  for (int i = 0; i < 8; i++) {
    int idx = tid + 256 * i, row = idx >> 6, c = idx & 63;
    u16 v = qp[(size_t)(b * NTc + t0 + row) * Dc + h * HDc + c];
    qt[row * QST + c] = f2bf(bf2f(v) * 0.125f);
  }
  wait_async0();
  __syncthreads();

  // ---- scores: each wave owns a 128-col strip of NS ----
  {
    int n0 = wave * 128;
    v16bf aq[2][2];
#pragma unroll
    for (int rt = 0; rt < 2; rt++)
#pragma unroll
      for (int ks = 0; ks < 2; ks++) aq[rt][ks] = ld_frag_a(qt, rt * 16, QST, ks * 32);

#pragma unroll
    for (int ct = 0; ct < 8; ct++) {
      int cb = n0 + ct * 16;
      v16bf b0 = ld_frag_b(kt, cb, KST, 0);
      v16bf b1 = ld_frag_b(kt, cb, KST, 32);
#pragma unroll
      for (int rt = 0; rt < 2; rt++) {
        v8f c = {};
        c = wmma_bf16(aq[rt][0], b0, c);
        c = wmma_bf16(aq[rt][1], b1, c);
#pragma unroll
        for (int i = 0; i < 8; i++)
          sc[(size_t)(rt * 16 + i + 8 * hh) * SST + cb + r] = c[i];
      }
    }
  }
  __syncthreads();

  // ---- softmax: each wave owns 4 rows; full-row reductions via shfl ----
#pragma unroll
  for (int rr = 0; rr < 4; rr++) {
    int row = wave * 4 + rr;
    float* srow = sc + (size_t)row * SST;
    const float* mrow = mask + (size_t)(t0 + row) * NSc;
    float mx = -3.0e38f;
    for (int i = 0; i < 32; i++) {
      float v = srow[lane + 32 * i] + mrow[lane + 32 * i];
      srow[lane + 32 * i] = v;
      mx = fmaxf(mx, v);
    }
    for (int off = 16; off; off >>= 1) mx = fmaxf(mx, __shfl_xor(mx, off, 32));
    float sum = 0.f;
    for (int i = 0; i < 32; i++) {
      float e = __expf(srow[lane + 32 * i] - mx);
      srow[lane + 32 * i] = e;
      sum += e;
    }
    for (int off = 16; off; off >>= 1) sum += __shfl_xor(sum, off, 32);
    float inv = 1.0f / sum;
    float* orow = wout + ((((size_t)b * Hc + h) * NTc + (t0 + row)) * NSc);
    u16* wrow = wt + (size_t)row * WST;  // kt no longer needed; alias is safe post-barrier
    for (int i = 0; i < 32; i++) {
      float w = srow[lane + 32 * i] * inv;
      orow[lane + 32 * i] = w;       // mandatory 256MB f32 output
      wrow[lane + 32 * i] = f2bf(w); // bf16 copy for P@V
    }
  }
  __syncthreads();

  // ---- ctx = P @ V : 32x64 output = 8 tiles, one per wave; K=NS in 32-steps ----
  {
    int rt = wave >> 2, cn = wave & 3;
    v8f cacc = {};
    for (int ks = 0; ks < 32; ks++) {
      int k0 = ks * 32;
      __syncthreads();  // previous vt consumers done
      {  // stage vt[e][sl] = vp[b, k0+sl, h*64+e]  (transpose on store)
        int e = tid & 63, sl0 = tid >> 6;
#pragma unroll
        for (int j = 0; j < 8; j++) {
          int sl = sl0 + 4 * j;
          vt[e * VST + sl] = vp[(size_t)(b * NSc + k0 + sl) * Dc + h * HDc + e];
        }
      }
      __syncthreads();
      v16bf aw = ld_frag_a(wt, rt * 16, WST, k0);
      v16bf bv = ld_frag_b(vt, cn * 16, VST, 0);
      cacc = wmma_bf16(aw, bv, cacc);
    }
#pragma unroll
    for (int i = 0; i < 8; i++) {
      int row = t0 + rt * 16 + i + 8 * hh;
      ctx[(size_t)(b * NTc + row) * Dc + h * HDc + cn * 16 + r] = f2bf(cacc[i]);
    }
  }
}

// ---------------------------------------------------------------------------
extern "C" void kernel_launch(void* const* d_in, const int* in_sizes, int n_in,
                              void* d_out, int out_size, void* d_ws, size_t ws_size,
                              hipStream_t stream) {
  (void)in_sizes; (void)n_in; (void)out_size; (void)ws_size;

  const float* q     = (const float*)d_in[0];
  const float* k     = (const float*)d_in[1];
  const float* v     = (const float*)d_in[2];
  const float* q_w   = (const float*)d_in[3];
  const float* k_w   = (const float*)d_in[4];
  const float* v_w   = (const float*)d_in[5];
  const float* out_w = (const float*)d_in[6];
  const float* b_q   = (const float*)d_in[7];
  const float* b_k   = (const float*)d_in[8];
  const float* b_v   = (const float*)d_in[9];
  const float* b_out = (const float*)d_in[10];
  const float* mask  = (const float*)d_in[11];

  const size_t MP = (size_t)Bc * NTc;  // 4096 rows
  u16* qp  = (u16*)d_ws;               // 3x bf16 projections + bf16 ctx = 32 MB ws
  u16* kp  = qp + MP * Dc;
  u16* vp  = kp + MP * Dc;
  u16* ctx = vp + MP * Dc;

  float* out     = (float*)d_out;      // [B, NT, D] f32
  float* weights = out + MP * Dc;      // [B, H, NT, NS] f32

  dim3 blk(256);
  dim3 gproj(Dc / 64, (int)(MP / 128));

  gemm_xwT<false, true><<<gproj, blk, 0, stream>>>(q, q_w, b_q, qp, (int)MP, Dc, Dc);
  gemm_xwT<false, true><<<gproj, blk, 0, stream>>>(k, k_w, b_k, kp, (int)MP, Dc, Dc);
  gemm_xwT<false, true><<<gproj, blk, 0, stream>>>(v, v_w, b_v, vp, (int)MP, Dc, Dc);

  (void)hipFuncSetAttribute((const void*)attn_fused,
                            hipFuncAttributeMaxDynamicSharedMemorySize,
                            (int)ATTN_SMEM);
  attn_fused<<<dim3(Bc * Hc * (NTc / TM)), blk, ATTN_SMEM, stream>>>(
      qp, kp, vp, mask, weights, ctx);

  gemm_xwT<true, false><<<gproj, blk, 0, stream>>>(ctx, out_w, b_out, out, (int)MP, Dc, Dc);
}
